// MLA_70557722738902
// MI455X (gfx1250) — compile-verified, matrix operand order checked by reading, not run on hardware
//
#include <hip/hip_runtime.h>
#include <cmath>

// ---------------------------------------------------------------------------
// MLA forward for gfx1250 (CDNA5, wave32, WMMA 16x16x32 bf16 + async-LDS DMA)
// ---------------------------------------------------------------------------

typedef __bf16 v16bf __attribute__((ext_vector_type(16)));
typedef __bf16 v8bf  __attribute__((ext_vector_type(8)));
typedef float  v8f   __attribute__((ext_vector_type(8)));

#define BATCH   2
#define SEQ     1024
#define TOK     (BATCH * SEQ)     // 2048 tokens
#define HIDDEN  3072
#define NHEADS  16
#define QLORA   1536
#define KVLORA  512
#define NOPE    128
#define ROPE    64
#define QKHD    (NOPE + ROPE)     // 192
#define VHD     128

// CDNA5 async global->LDS copy (16 bytes per lane), tracked by ASYNCcnt.
// vdst = per-lane LDS byte address (low 32 bits of generic LDS pointer),
// vaddr = per-lane 64-bit global address.
#define ASYNC_COPY_B128(gptr, lptr)                                          \
    asm volatile("global_load_async_to_lds_b128 %0, %1, off"                 \
                 :: "v"((unsigned)(size_t)(lptr)),                           \
                    "v"((unsigned long long)(size_t)(gptr))                  \
                 : "memory")

#define WAIT_ASYNCCNT0() asm volatile("s_wait_asynccnt 0" ::: "memory")

// Load a bf16 WMMA fragment: 8 contiguous elements at p, 8 more at p+16.
// Matches ISA 16-bit A-matrix 16x32 layout when p = row*stride + kstep + hb8
// (hb8 = 0 for lanes 0-15, 8 for lanes 16-31).
static __device__ __forceinline__ v16bf load_frag(const __bf16* p) {
    union { v16bf v; v8bf h[2]; } u;
    u.h[0] = *(const v8bf*)(p);
    u.h[1] = *(const v8bf*)(p + 16);
    return u.v;
}

// ---------------------------------------------------------------------------
// fp32 -> bf16 conversion (grid-stride)
// ---------------------------------------------------------------------------
__global__ void f32_to_bf16(const float* __restrict__ x, __bf16* __restrict__ y, long n) {
    long i = (long)blockIdx.x * blockDim.x + threadIdx.x;
    long stride = (long)gridDim.x * blockDim.x;
    for (; i < n; i += stride) y[i] = (__bf16)x[i];
}

// ---------------------------------------------------------------------------
// RMSNorm over first C columns of rows with stride ld, write bf16 [rows, C]
// ---------------------------------------------------------------------------
__global__ __launch_bounds__(256) void rmsnorm_to_bf16(
    const float* __restrict__ x, int ld,
    const float* __restrict__ w, __bf16* __restrict__ y, int C) {
    const int row = blockIdx.x;
    const float* xr = x + (size_t)row * ld;
    float ss = 0.f;
    for (int c = threadIdx.x; c < C; c += 256) { float v = xr[c]; ss += v * v; }
    __shared__ float red[256];
    red[threadIdx.x] = ss;
    __syncthreads();
    for (int s = 128; s > 0; s >>= 1) {
        if (threadIdx.x < s) red[threadIdx.x] += red[threadIdx.x + s];
        __syncthreads();
    }
    float inv = rsqrtf(red[0] / (float)C + 1e-6f);
    __bf16* yr = y + (size_t)row * C;
    for (int c = threadIdx.x; c < C; c += 256) yr[c] = (__bf16)(w[c] * xr[c] * inv);
}

// ---------------------------------------------------------------------------
// GEMM: C[M,N] = A[M,K] * B[N,K]^T + bias[N], bf16 inputs, f32 out.
// 256 threads (8 waves); 128x128 tile; each wave owns a 16x128 strip.
// Global->LDS staging uses CDNA5 async-LDS DMA; B fragments are preloaded so
// the 8-WMMA chain issues back-to-back behind a single DS wait.
// K multiple of 32. N guarded; M assumed multiple of 128 (always 2048 here).
// ---------------------------------------------------------------------------
__global__ __launch_bounds__(256) void gemm_bf16(
    const __bf16* __restrict__ A, const __bf16* __restrict__ Bw,
    const float* __restrict__ bias, float* __restrict__ C,
    int M, int N, int K) {
    __shared__ __bf16 As[128 * 32];
    __shared__ __bf16 Bs[128 * 32];

    const int tid  = threadIdx.x;
    const int lane = tid & 31;
    const int wave = tid >> 5;
    const int lr   = lane & 15;
    const int hb8  = (lane < 16) ? 0 : 8;   // K-offset for A/B frags, row offset for C

    const int m0 = blockIdx.y * 128;
    const int n0 = blockIdx.x * 128;

    v8f acc[8];
    for (int t = 0; t < 8; ++t) {
        int col = n0 + t * 16 + lr;
        float bv = (col < N) ? bias[col] : 0.f;
        for (int e = 0; e < 8; ++e) acc[t][e] = bv;
    }

    const int lrow  = tid >> 1;          // 0..127
    const int lhalf = (tid & 1) * 16;    // 0 or 16 (elements)
    const bool bok  = (n0 + lrow) < N;

    // zero-fill out-of-range B rows once (async path skips them)
    if (!bok) {
        uint4 z{};
        *(uint4*)(&Bs[lrow * 32 + lhalf])     = z;
        *(uint4*)(&Bs[lrow * 32 + lhalf + 8]) = z;
    }

    const __bf16* ga = A  + (size_t)(m0 + lrow) * K + lhalf;
    const __bf16* gb = Bw + (size_t)(n0 + lrow) * K + lhalf;

    for (int k0 = 0; k0 < K; k0 += 32) {
        // ---- async DMA: global -> LDS, 32 bytes of A and of B per thread ----
        ASYNC_COPY_B128(ga + k0,     &As[lrow * 32 + lhalf]);
        ASYNC_COPY_B128(ga + k0 + 8, &As[lrow * 32 + lhalf + 8]);
        if (bok) {
            ASYNC_COPY_B128(gb + k0,     &Bs[lrow * 32 + lhalf]);
            ASYNC_COPY_B128(gb + k0 + 8, &Bs[lrow * 32 + lhalf + 8]);
        }
        if (k0 + 32 < K)
            __builtin_prefetch(ga + k0 + 32, 0, 0);
        WAIT_ASYNCCNT0();
        __syncthreads();

        // ---- preload all fragments, then an uninterrupted 8-WMMA chain ----
        v16bf af = load_frag(&As[(wave * 16 + lr) * 32 + hb8]);
        v16bf bfr[8];
        for (int t = 0; t < 8; ++t)
            bfr[t] = load_frag(&Bs[(t * 16 + lr) * 32 + hb8]);
        for (int t = 0; t < 8; ++t)
            acc[t] = __builtin_amdgcn_wmma_f32_16x16x32_bf16(
                false, af, false, bfr[t], (short)0, acc[t], false, false);
        __syncthreads();
    }

    for (int t = 0; t < 8; ++t) {
        int col = n0 + t * 16 + lr;
        if (col >= N) continue;
        for (int e = 0; e < 8; ++e) {
            int row = m0 + wave * 16 + e + hb8;
            if (row < M) C[(size_t)row * N + col] = acc[t][e];
        }
    }
}

// ---------------------------------------------------------------------------
// Pack + RoPE: build WMMA-friendly Q/K [B,H,S,192] bf16 and V^T [B,H,128,S].
// ---------------------------------------------------------------------------
__global__ __launch_bounds__(256) void pack_qkv(
    const float* __restrict__ qf,    // [B,S,16,192]
    const float* __restrict__ kva,   // [B,S,576]  (last 64 = k_rope, pre-rope)
    const float* __restrict__ kvb,   // [B,S,16,256] (128 k_nope | 128 v)
    __bf16* __restrict__ qbf,        // [B,16,S,192]
    __bf16* __restrict__ kbf,        // [B,16,S,192]
    __bf16* __restrict__ vbfT) {     // [B,16,128,S]
    const int s = blockIdx.x, b = blockIdx.y;
    const size_t tok = (size_t)b * SEQ + s;

    for (int i = threadIdx.x; i < NHEADS * QKHD; i += 256) {
        int h = i / QKHD, d = i % QKHD;
        size_t dst = ((size_t)(b * NHEADS + h) * SEQ + s) * QKHD + d;
        float qv, kv;
        if (d < NOPE) {
            qv = qf[tok * (NHEADS * QKHD) + h * QKHD + d];
            kv = kvb[tok * (NHEADS * 256) + h * 256 + d];
        } else {
            int p2 = d - NOPE;
            int pair = p2 >> 1;
            float invf = __powf(10000.f, -(float)(2 * pair) / (float)ROPE);
            float sn, cs;
            __sincosf((float)s * invf, &sn, &cs);
            float q1 = qf[tok * (NHEADS * QKHD) + h * QKHD + NOPE + 2 * pair];
            float q2 = qf[tok * (NHEADS * QKHD) + h * QKHD + NOPE + 2 * pair + 1];
            float k1 = kva[tok * (KVLORA + ROPE) + KVLORA + 2 * pair];
            float k2 = kva[tok * (KVLORA + ROPE) + KVLORA + 2 * pair + 1];
            if ((p2 & 1) == 0) { qv = q1 * cs - q2 * sn; kv = k1 * cs - k2 * sn; }
            else               { qv = q1 * sn + q2 * cs; kv = k1 * sn + k2 * cs; }
        }
        qbf[dst] = (__bf16)qv;
        kbf[dst] = (__bf16)kv;
    }
    for (int i = threadIdx.x; i < NHEADS * VHD; i += 256) {
        int h = i / VHD, d = i % VHD;
        vbfT[((size_t)(b * NHEADS + h) * VHD + d) * SEQ + s] =
            (__bf16)kvb[tok * (NHEADS * 256) + h * 256 + NOPE + d];
    }
}

// ---------------------------------------------------------------------------
// Flash attention (causal): grid (S/64, B*H), 128 threads = 4 waves.
// Each wave: 16 query rows, full 128-wide V accumulation, online softmax.
// QK^T: 24 WMMA / key block; PV: 16 WMMA / key block.
// ---------------------------------------------------------------------------
__global__ __launch_bounds__(128) void flash_attn(
    const __bf16* __restrict__ qbf,   // [B*H, S, 192]
    const __bf16* __restrict__ kbf,   // [B*H, S, 192]
    const __bf16* __restrict__ vbfT,  // [B*H, 128, S]
    __bf16* __restrict__ obf,         // [B, S, H*128]
    float scale) {
    __shared__ __bf16 Plds[4][16 * 64];   // per-wave P tile (16 rows x 64 keys)

    const int lane = threadIdx.x & 31;
    const int wave = threadIdx.x >> 5;
    const int lr   = lane & 15;
    const int hb8  = (lane < 16) ? 0 : 8;

    const int bh  = blockIdx.y;
    const int qb0 = blockIdx.x * 64;
    const int qw  = qb0 + wave * 16;      // wave's query-row base

    const __bf16* Q  = qbf  + (size_t)bh * SEQ * QKHD;
    const __bf16* Kp = kbf  + (size_t)bh * SEQ * QKHD;
    const __bf16* Vp = vbfT + (size_t)bh * VHD * SEQ;

    // Q strip resident in registers: 6 A-fragments (16x192)
    v16bf qfrag[6];
    {
        const __bf16* qrow = Q + (size_t)(qw + lr) * QKHD;
        for (int ks = 0; ks < 6; ++ks)
            qfrag[ks] = load_frag(qrow + ks * 32 + hb8);
    }

    v8f acc[8];
    for (int n = 0; n < 8; ++n)
        for (int e = 0; e < 8; ++e) acc[n][e] = 0.f;
    float m_i[8], l_i[8];
    for (int e = 0; e < 8; ++e) { m_i[e] = -3.0e38f; l_i[e] = 0.f; }

    const int nblocks = blockIdx.x + 1;   // causal: key blocks 0..qblock
    for (int jb = 0; jb < nblocks; ++jb) {
        const int kb0 = jb * 64;

        // ---- S = Q K^T for this 16x64 tile ----
        v8f st[4];
        for (int t = 0; t < 4; ++t)
            for (int e = 0; e < 8; ++e) st[t][e] = 0.f;
        for (int ks = 0; ks < 6; ++ks) {
            v16bf kf[4];
            for (int t = 0; t < 4; ++t)
                kf[t] = load_frag(Kp + (size_t)(kb0 + t * 16 + lr) * QKHD + ks * 32 + hb8);
            for (int t = 0; t < 4; ++t)
                st[t] = __builtin_amdgcn_wmma_f32_16x16x32_bf16(
                    false, qfrag[ks], false, kf[t], (short)0, st[t], false, false);
        }

        // ---- online softmax (per row e; 16 lanes share a row) ----
        const bool diag = (jb == blockIdx.x);
        for (int e = 0; e < 8; ++e) {
            int qrow = qw + e + hb8;
            float sv[4];
            float mx = -3.0e38f;
            for (int t = 0; t < 4; ++t) {
                int kcol = kb0 + t * 16 + lr;
                float s = st[t][e] * scale;
                if (diag && kcol > qrow) s = -1.0e9f;
                sv[t] = s;
                mx = fmaxf(mx, s);
            }
            for (int off = 1; off < 16; off <<= 1)
                mx = fmaxf(mx, __shfl_xor(mx, off, 32));
            float mnew  = fmaxf(m_i[e], mx);
            float alpha = __expf(m_i[e] - mnew);
            float rs = 0.f;
            for (int t = 0; t < 4; ++t) {
                float p = __expf(sv[t] - mnew);
                rs += p;
                Plds[wave][(e + hb8) * 64 + t * 16 + lr] = (__bf16)p;
            }
            for (int off = 1; off < 16; off <<= 1)
                rs += __shfl_xor(rs, off, 32);
            l_i[e] = l_i[e] * alpha + rs;
            m_i[e] = mnew;
            for (int n = 0; n < 8; ++n) acc[n][e] *= alpha;
        }

        // ---- O += P V  (P from LDS in A-fragment layout, V^T contiguous) ----
        for (int kk = 0; kk < 2; ++kk) {
            v16bf pf = load_frag(&Plds[wave][lr * 64 + kk * 32 + hb8]);
            v16bf vf[8];
            for (int n = 0; n < 8; ++n)
                vf[n] = load_frag(Vp + (size_t)(n * 16 + lr) * SEQ + kb0 + kk * 32 + hb8);
            for (int n = 0; n < 8; ++n)
                acc[n] = __builtin_amdgcn_wmma_f32_16x16x32_bf16(
                    false, pf, false, vf[n], (short)0, acc[n], false, false);
        }
    }

    // ---- epilogue: normalize and store bf16 [B, S, H*128] ----
    const int b = bh >> 4, h = bh & 15;
    for (int e = 0; e < 8; ++e) {
        float inv = 1.0f / l_i[e];
        int srow = qw + e + hb8;
        __bf16* orow = obf + ((size_t)(b * SEQ + srow)) * (NHEADS * VHD) + h * VHD;
        for (int n = 0; n < 8; ++n)
            orow[n * 16 + lr] = (__bf16)(acc[n][e] * inv);
    }
}

// ---------------------------------------------------------------------------
// Host launch
// ---------------------------------------------------------------------------
extern "C" void kernel_launch(void* const* d_in, const int* in_sizes, int n_in,
                              void* d_out, int out_size, void* d_ws, size_t ws_size,
                              hipStream_t stream) {
    (void)in_sizes; (void)n_in; (void)out_size; (void)ws_size;

    const float* hidden = (const float*)d_in[0];
    // d_in[1] = mask (causal applied analytically)
    const float* wqa  = (const float*)d_in[2];
    const float* bqa  = (const float*)d_in[3];
    const float* qnw  = (const float*)d_in[4];
    const float* wqb  = (const float*)d_in[5];
    const float* bqb  = (const float*)d_in[6];
    const float* wkva = (const float*)d_in[7];
    const float* bkva = (const float*)d_in[8];
    const float* kvnw = (const float*)d_in[9];
    const float* wkvb = (const float*)d_in[10];
    const float* bkvb = (const float*)d_in[11];
    const float* wo   = (const float*)d_in[12];
    const float* bo   = (const float*)d_in[13];
    float* out = (float*)d_out;

    // ---- workspace layout ----
    char* base = (char*)d_ws;
    size_t off = 0;
    auto alloc = [&](size_t bytes) -> char* {
        char* p = base + off;
        off += (bytes + 255) & ~(size_t)255;
        return p;
    };
    __bf16* hbf      = (__bf16*)alloc((size_t)TOK * HIDDEN * 2);
    __bf16* wbuf     = (__bf16*)alloc((size_t)HIDDEN * TOK * 2);     // max weight: wo (3072x2048)
    float*  qlat_f   = (float*) alloc((size_t)TOK * QLORA * 4);      // reused later as obf
    __bf16* qlatn_bf = (__bf16*)alloc((size_t)TOK * QLORA * 2);
    float*  q_f      = (float*) alloc((size_t)TOK * HIDDEN * 4);
    float*  kva_f    = (float*) alloc((size_t)TOK * (KVLORA + ROPE) * 4);
    __bf16* kvn_bf   = (__bf16*)alloc((size_t)TOK * KVLORA * 2);
    float*  kvb_f    = (float*) alloc((size_t)TOK * NHEADS * 256 * 4);
    __bf16* qbf      = (__bf16*)alloc((size_t)BATCH * NHEADS * SEQ * QKHD * 2);
    __bf16* kbf      = (__bf16*)alloc((size_t)BATCH * NHEADS * SEQ * QKHD * 2);
    __bf16* vbfT     = (__bf16*)alloc((size_t)BATCH * NHEADS * VHD * SEQ * 2);
    __bf16* obf      = (__bf16*)qlat_f;   // alias: qlat_f dead after q rmsnorm

    auto cvt = [&](const float* src, __bf16* dst, long n) {
        f32_to_bf16<<<4096, 256, 0, stream>>>(src, dst, n);
    };

    // 1) hidden -> bf16
    cvt(hidden, hbf, (long)TOK * HIDDEN);

    // 2) q_lat = hidden @ wq_a^T + b
    cvt(wqa, wbuf, (long)QLORA * HIDDEN);
    gemm_bf16<<<dim3(QLORA / 128, TOK / 128), 256, 0, stream>>>(
        hbf, wbuf, bqa, qlat_f, TOK, QLORA, HIDDEN);

    // 3) rms_norm(q_lat) -> bf16
    rmsnorm_to_bf16<<<TOK, 256, 0, stream>>>(qlat_f, QLORA, qnw, qlatn_bf, QLORA);

    // 4) q = qlat_norm @ wq_b^T + b
    cvt(wqb, wbuf, (long)HIDDEN * QLORA);
    gemm_bf16<<<dim3(HIDDEN / 128, TOK / 128), 256, 0, stream>>>(
        qlatn_bf, wbuf, bqb, q_f, TOK, HIDDEN, QLORA);

    // 5) kv_a = hidden @ wkv_a^T + b   (N=576, guarded)
    cvt(wkva, wbuf, (long)(KVLORA + ROPE) * HIDDEN);
    gemm_bf16<<<dim3((KVLORA + ROPE + 127) / 128, TOK / 128), 256, 0, stream>>>(
        hbf, wbuf, bkva, kva_f, TOK, KVLORA + ROPE, HIDDEN);

    // 6) rms_norm(kv_lat[:512]) -> bf16
    rmsnorm_to_bf16<<<TOK, 256, 0, stream>>>(kva_f, KVLORA + ROPE, kvnw, kvn_bf, KVLORA);

    // 7) kv = kv_norm @ wkv_b^T + b
    cvt(wkvb, wbuf, (long)(NHEADS * 256) * KVLORA);
    gemm_bf16<<<dim3(NHEADS * 256 / 128, TOK / 128), 256, 0, stream>>>(
        kvn_bf, wbuf, bkvb, kvb_f, TOK, NHEADS * 256, KVLORA);

    // 8) rope + pack Q/K/V into WMMA-friendly bf16 layouts
    pack_qkv<<<dim3(SEQ, BATCH), 256, 0, stream>>>(q_f, kva_f, kvb_f, qbf, kbf, vbfT);

    // 9) causal flash attention -> obf [B,S,H*128] bf16
    const float scale = 1.0f / sqrtf((float)QKHD);
    flash_attn<<<dim3(SEQ / 64, BATCH * NHEADS), 128, 0, stream>>>(
        qbf, kbf, vbfT, obf, scale);

    // 10) out = attn_out @ wo^T + b
    cvt(wo, wbuf, (long)HIDDEN * (NHEADS * VHD));
    gemm_bf16<<<dim3(HIDDEN / 128, TOK / 128), 256, 0, stream>>>(
        obf, wbuf, bo, out, TOK, HIDDEN, NHEADS * VHD);
}
